// CrossAttention_47553877901587
// MI455X (gfx1250) — matrix operand order repaired and checked by reading, not compiled, not run on hardware
//
#include <hip/hip_runtime.h>

// ---------------------------------------------------------------------------
// CrossAttention for MI455X (gfx1250): all matmuls via v_wmma_f32_16x16x32_bf16
//  - weights pre-swizzled once into B-fragment-major bf16 (2x b128 per frag)
//  - GEMM: one 16x128 C-strip per wave (8 WMMAs reuse one A fragment / k-step)
//  - fused flash-style attention with LDS P re-layout
// ---------------------------------------------------------------------------

typedef __attribute__((ext_vector_type(16))) __bf16 v16bf;
typedef __attribute__((ext_vector_type(8)))  float  v8f;

union BfFrag {               // assemble WMMA operands directly from dwords
  v16bf        v;
  unsigned int u[8];
};

__device__ __forceinline__ unsigned short f2bf_bits(float f) {
  unsigned int u = __builtin_bit_cast(unsigned int, f);
  u += 0x7FFFu + ((u >> 16) & 1u);           // round-to-nearest-even
  return (unsigned short)(u >> 16);
}
__device__ __forceinline__ unsigned int pack_bf16(float lo, float hi) {
  return (unsigned int)f2bf_bits(lo) | ((unsigned int)f2bf_bits(hi) << 16);
}

// ---------------------------------------------------------------------------
// One-time weight swizzle: W[K,512] f32 -> Wb dwords laid out [K/32][512][2][8]
//   Wb[((s*512+n)*2+h)*8+v] = pack(W[32s+2v+16h, n], W[32s+2v+16h+1, n])
// so a lane's whole B fragment is 8 consecutive dwords (two b128 loads).
// ---------------------------------------------------------------------------
__global__ void swizzle_weights(const float* __restrict__ W,
                                unsigned int* __restrict__ Wb, int K) {
  constexpr int N = 512;
  const int idx = blockIdx.x * blockDim.x + threadIdx.x;   // dword index
  if (idx >= (K >> 5) * N * 16) return;
  const int v = idx & 7;
  const int h = (idx >> 3) & 1;
  const int n = (idx >> 4) & (N - 1);
  const int s = idx >> 13;
  const int kb = s * 32 + 2 * v + 16 * h;
  Wb[idx] = pack_bf16(W[(size_t)kb * N + n], W[(size_t)(kb + 1) * N + n]);
}

// ---------------------------------------------------------------------------
// GEMM: C[M,512] = A[M,K] @ W[K,512] + bias (W pre-swizzled bf16 fragments).
// Grid: (ceil(M/16)), block 128 = 4 waves; wave w covers columns [128w,128w+128).
// ---------------------------------------------------------------------------
template <int A_BF16, int OUT_F32>
__global__ void gemm_wmma_bf16(const void* __restrict__ Aptr,
                               const unsigned int* __restrict__ Wb,
                               const float* __restrict__ bias,
                               void* __restrict__ Cptr,
                               int M, int K) {
  constexpr int N = 512;
  const int lane = threadIdx.x & 31;
  const int wave = threadIdx.x >> 5;
  const int half = lane >> 4;
  const int l16  = lane & 15;
  const int mt   = blockIdx.x;
  const int n0w  = wave * 128;                 // this wave's column strip
  const int mA   = mt * 16 + l16;              // row fed into A fragment
  const int mAr  = (mA < M) ? mA : (M - 1);    // clamped (branchless guard)
  const bool mok = (mA < M);

  const float*          Af = (const float*)Aptr;
  const unsigned short* Ab = (const unsigned short*)Aptr;

  v8f acc[8];
#pragma unroll
  for (int t = 0; t < 8; ++t) acc[t] = (v8f){};

  for (int kk = 0; kk < K; kk += 32) {
    const int s = kk >> 5;
    // ---- A fragment (16x32 bf16), loaded once, reused for 8 WMMAs
    BfFrag a;
#pragma unroll
    for (int v = 0; v < 8; ++v) {
      const int k0 = kk + 2 * v + 16 * half;
      if (A_BF16) {
        unsigned int w = *(const unsigned int*)(Ab + (size_t)mAr * K + k0);
        a.u[v] = mok ? w : 0u;
      } else {
        const float a0 = Af[(size_t)mAr * K + k0];
        const float a1 = Af[(size_t)mAr * K + k0 + 1];
        a.u[v] = mok ? pack_bf16(a0, a1) : 0u;
      }
    }
    if (!A_BF16 && kk + 32 < K)
      __builtin_prefetch(Af + (size_t)mAr * K + kk + 32, 0, 1);

    // ---- 8 n-tiles: B fragment = 8 consecutive dwords, then WMMA
#pragma unroll
    for (int t = 0; t < 8; ++t) {
      const unsigned int* bp =
          Wb + ((size_t)(s * N + n0w + t * 16 + l16) << 4) + (half << 3);
      BfFrag b;
#pragma unroll
      for (int v = 0; v < 8; ++v) b.u[v] = bp[v];
      acc[t] = __builtin_amdgcn_wmma_f32_16x16x32_bf16(
          false, a.v, false, b.v, (short)0, acc[t], false, false);
    }
  }

#pragma unroll
  for (int t = 0; t < 8; ++t) {
    const int nB       = n0w + t * 16 + l16;
    const float bias_n = bias[nB];
#pragma unroll
    for (int v = 0; v < 8; ++v) {
      const int mm = mt * 16 + v + 8 * half;   // C layout: m = v + 8*half
      if (mm < M) {
        const float val = acc[t][v] + bias_n;
        if (OUT_F32)
          ((float*)Cptr)[(size_t)mm * N + nB] = val;
        else
          ((unsigned short*)Cptr)[(size_t)mm * N + nB] = f2bf_bits(val);
      }
    }
  }
}

// ---------------------------------------------------------------------------
// Fused attention: per (b,h) and 16-row q-tile (one wave), flash-style.
//   S = Q Kt / 8 (SKV=77 padded to 80: 5 n-tiles), softmax over keys,
//   P staged via LDS (C-layout -> A-fragment re-layout), O = P V (K padded 96).
// Grid: (B*H, SQ/64), block 128 (4 waves, each a 16-row q tile).
// ---------------------------------------------------------------------------
__global__ void attn_wmma_bf16(const unsigned short* __restrict__ Qb,
                               const unsigned short* __restrict__ Kb,
                               const unsigned short* __restrict__ Vb,
                               unsigned short* __restrict__ Ob) {
  __shared__ unsigned short ldsP[4][16][96];   // per-wave P tile, bf16, K pad 96

  const int lane = threadIdx.x & 31;
  const int wave = threadIdx.x >> 5;
  const int half = lane >> 4;
  const int l16  = lane & 15;
  const int b    = blockIdx.x >> 3;
  const int h    = blockIdx.x & 7;
  const int qr   = (blockIdx.y * 4 + wave) * 16;   // q row base (< 4096)

  // ---- Q A-fragments: 16x64, two 32-wide K steps (pairs contiguous in d)
  BfFrag qa[2];
  const unsigned short* qrow =
      Qb + ((size_t)(b * 4096 + qr + l16) * 512 + h * 64);
#pragma unroll
  for (int s = 0; s < 2; ++s)
#pragma unroll
    for (int v = 0; v < 8; ++v) {
      const int k0 = s * 32 + 2 * v + 16 * half;
      qa[s].u[v] = *(const unsigned int*)(qrow + k0);
    }

  // ---- scores S = Q Kt : 5 key tiles of 16 (clamped, branchless padding)
  v8f accs[5];
#pragma unroll
  for (int t = 0; t < 5; ++t) {
    v8f acc = {};
    const int  n   = t * 16 + l16;             // key index (column)
    const int  nr  = (n < 77) ? n : 76;
    const bool nok = (n < 77);
    const unsigned short* krow = Kb + ((size_t)(b * 77 + nr) * 512 + h * 64);
#pragma unroll
    for (int s = 0; s < 2; ++s) {
      BfFrag bk;
#pragma unroll
      for (int v = 0; v < 8; ++v) {
        const int k0 = s * 32 + 2 * v + 16 * half;  // d index
        const unsigned int w = *(const unsigned int*)(krow + k0);
        bk.u[v] = nok ? w : 0u;
      }
      acc = __builtin_amdgcn_wmma_f32_16x16x32_bf16(
          false, qa[s].v, false, bk.v, (short)0, acc, false, false);
    }
    accs[t] = acc;
  }

  // ---- softmax over keys. Row m = v + 8*half lives in one 16-lane half.
  const float scale = 0.125f;                       // 1/sqrt(64)
  float rsum[8];
#pragma unroll
  for (int v = 0; v < 8; ++v) {
    float m = -1e30f;
#pragma unroll
    for (int t = 0; t < 5; ++t) {
      const int n = t * 16 + l16;
      float s = accs[t][v] * scale;
      s = (n < 77) ? s : -1e30f;
      accs[t][v] = s;
      m = fmaxf(m, s);
    }
    m = fmaxf(m, __shfl_xor(m, 1, 32));
    m = fmaxf(m, __shfl_xor(m, 2, 32));
    m = fmaxf(m, __shfl_xor(m, 4, 32));
    m = fmaxf(m, __shfl_xor(m, 8, 32));
    float sum = 0.0f;
#pragma unroll
    for (int t = 0; t < 5; ++t) {
      const float p = __expf(accs[t][v] - m);       // masked -> 0
      accs[t][v] = p;
      sum += p;
    }
    sum += __shfl_xor(sum, 1, 32);
    sum += __shfl_xor(sum, 2, 32);
    sum += __shfl_xor(sum, 4, 32);
    sum += __shfl_xor(sum, 8, 32);
    rsum[v] = sum;
  }

  // ---- stage P (C layout) to LDS as bf16, pad K dim 80..95 with zeros
#pragma unroll
  for (int t = 0; t < 5; ++t)
#pragma unroll
    for (int v = 0; v < 8; ++v)
      ldsP[wave][v + 8 * half][t * 16 + l16] = f2bf_bits(accs[t][v]);
#pragma unroll
  for (int v = 0; v < 8; ++v)
    ldsP[wave][v + 8 * half][80 + l16] = 0;
  __syncthreads();

  // ---- reload P as A-fragments (3 K steps of 32)
  BfFrag pa[3];
  const unsigned int* prow = (const unsigned int*)&ldsP[wave][l16][0];
#pragma unroll
  for (int s = 0; s < 3; ++s)
#pragma unroll
    for (int v = 0; v < 8; ++v) {
      const int k0 = s * 32 + 2 * v + 16 * half;
      pa[s].u[v] = prow[k0 >> 1];
    }

  // ---- O = P V : 4 d-tiles of 16, normalize by row sum, store bf16
#pragma unroll
  for (int t2 = 0; t2 < 4; ++t2) {
    v8f acc = {};
    const int col = h * 64 + t2 * 16 + l16;         // d index (column)
#pragma unroll
    for (int s = 0; s < 3; ++s) {
      BfFrag vf;
#pragma unroll
      for (int v = 0; v < 8; ++v) {
        const int k0  = s * 32 + 2 * v + 16 * half; // key index
        const int r0  = (k0 < 77) ? k0 : 76;
        const int r1  = (k0 + 1 < 77) ? (k0 + 1) : 76;
        const unsigned int e0 =
            (k0 < 77) ? (unsigned int)Vb[(size_t)(b * 77 + r0) * 512 + col] : 0u;
        const unsigned int e1 =
            (k0 + 1 < 77) ? (unsigned int)Vb[(size_t)(b * 77 + r1) * 512 + col]
                          : 0u;
        vf.u[v] = e0 | (e1 << 16);
      }
      acc = __builtin_amdgcn_wmma_f32_16x16x32_bf16(
          false, pa[s].v, false, vf.v, (short)0, acc, false, false);
    }
#pragma unroll
    for (int v = 0; v < 8; ++v) {
      const float o  = acc[v] / rsum[v];
      const int   mm = qr + v + 8 * half;
      Ob[(size_t)(b * 4096 + mm) * 512 + h * 64 + t2 * 16 + l16] =
          f2bf_bits(o);
    }
  }
}

// ---------------------------------------------------------------------------
// Launch: swizzle weights -> Q/K/V projections -> attention -> O projection.
// ---------------------------------------------------------------------------
extern "C" void kernel_launch(void* const* d_in, const int* in_sizes, int n_in,
                              void* d_out, int out_size, void* d_ws,
                              size_t ws_size, hipStream_t stream) {
  const float* latent  = (const float*)d_in[0];  // [8,4096,512]
  const float* context = (const float*)d_in[1];  // [8,77,768]
  const float* wq = (const float*)d_in[2];
  const float* bq = (const float*)d_in[3];
  const float* wk = (const float*)d_in[4];
  const float* bk = (const float*)d_in[5];
  const float* wv = (const float*)d_in[6];
  const float* bv = (const float*)d_in[7];
  const float* wo = (const float*)d_in[8];
  const float* bo = (const float*)d_in[9];
  float* out = (float*)d_out;                    // [8,4096,512] f32

  char* ws = (char*)d_ws;
  unsigned short* Qb  = (unsigned short*)(ws);              // 32768*512 bf16
  unsigned short* Kb  = (unsigned short*)(ws + 33554432);   //   616*512 bf16
  unsigned short* Vb  = (unsigned short*)(ws + 34185216);   //   616*512 bf16
  unsigned short* Obf = (unsigned short*)(ws + 34816000);   // 32768*512 bf16
  unsigned int*   Wq  = (unsigned int*)(ws + 68370432);     // 512*512/2 dwords
  unsigned int*   Wk  = (unsigned int*)(ws + 68894720);     // 768*512/2 dwords
  unsigned int*   Wv  = (unsigned int*)(ws + 69681152);     // 768*512/2 dwords
  unsigned int*   Wo  = (unsigned int*)(ws + 70467584);     // 512*512/2 dwords

  const dim3 blk(128);

  // one-time weight swizzle into B-fragment-major bf16
  swizzle_weights<<<dim3(512), dim3(256), 0, stream>>>(wq, Wq, 512);
  swizzle_weights<<<dim3(768), dim3(256), 0, stream>>>(wk, Wk, 768);
  swizzle_weights<<<dim3(768), dim3(256), 0, stream>>>(wv, Wv, 768);
  swizzle_weights<<<dim3(512), dim3(256), 0, stream>>>(wo, Wo, 512);

  // Q = latent @ wq + bq          (M=32768, K=512) -> bf16
  gemm_wmma_bf16<0, 0><<<dim3(2048), blk, 0, stream>>>(
      latent, Wq, bq, Qb, 32768, 512);
  // K,V = context @ w{k,v} + b    (M=616, K=768) -> bf16
  gemm_wmma_bf16<0, 0><<<dim3(39), blk, 0, stream>>>(
      context, Wk, bk, Kb, 616, 768);
  gemm_wmma_bf16<0, 0><<<dim3(39), blk, 0, stream>>>(
      context, Wv, bv, Vb, 616, 768);
  // fused softmax(QKt/8) V        grid (B*H=64, 4096/64=64)
  attn_wmma_bf16<<<dim3(64, 64), blk, 0, stream>>>(Qb, Kb, Vb, Obf);
  // out = attn @ wo + bo          (bf16 A, f32 out)
  gemm_wmma_bf16<1, 1><<<dim3(2048), blk, 0, stream>>>(
      Obf, Wo, bo, out, 32768, 512);
}
